// cnn_kan_with_attention_49632642073240
// MI455X (gfx1250) — matrix operand order repaired and checked by reading, not compiled, not run on hardware
//
#include <hip/hip_runtime.h>
#include <hip/hip_bf16.h>

typedef __attribute__((ext_vector_type(8)))  float          v8f;
typedef __attribute__((ext_vector_type(4)))  float          v4f;
typedef __attribute__((ext_vector_type(16))) __bf16         v16bf;
typedef __attribute__((ext_vector_type(16))) unsigned short v16u;
typedef __attribute__((ext_vector_type(4)))  unsigned int   v4u;
typedef __attribute__((ext_vector_type(4)))  int            v4i;
typedef __attribute__((ext_vector_type(8)))  int            v8i;

union U16 { v16u u; v16bf b; v4u q[2]; };

#define KCHUNK 512   // f32 elements per tile row staged by the TDM

__device__ __forceinline__ unsigned short f2bf(float f) {
    unsigned int u = __float_as_uint(f);
    u = (u + 0x7FFFu + ((u >> 16) & 1u)) >> 16;   // RNE
    return (unsigned short)u;
}

__device__ __forceinline__ float selu_f(float x) {
    const float a = 1.6732632423543772f, s = 1.0507009873554805f;
    return s * (x > 0.0f ? x : a * (expf(x) - 1.0f));
}

__device__ __forceinline__ void bspline8(float x, float* out) {
    const float h = 0.4f;
    float t[12];
#pragma unroll
    for (int j = 0; j < 12; ++j) t[j] = (float)(j - 3) * h - 1.0f;
    float b0[11];
#pragma unroll
    for (int j = 0; j < 11; ++j) b0[j] = (x >= t[j] && x < t[j + 1]) ? 1.0f : 0.0f;
    float b1[10];
#pragma unroll
    for (int j = 0; j < 10; ++j)
        b1[j] = (x - t[j]) / (t[j + 1] - t[j]) * b0[j] + (t[j + 2] - x) / (t[j + 2] - t[j + 1]) * b0[j + 1];
    float b2[9];
#pragma unroll
    for (int j = 0; j < 9; ++j)
        b2[j] = (x - t[j]) / (t[j + 2] - t[j]) * b1[j] + (t[j + 3] - x) / (t[j + 3] - t[j + 1]) * b1[j + 1];
#pragma unroll
    for (int j = 0; j < 8; ++j)
        out[j] = (x - t[j]) / (t[j + 3] - t[j]) * b2[j] + (t[j + 4] - x) / (t[j + 4] - t[j + 1]) * b2[j + 1];
}

// A-fragment: 16x32 bf16, contiguous k order -> two aligned 16B runs at ap[0..7], ap[8..15]
__device__ __forceinline__ void loadA(U16& Af, const unsigned short* ap, bool valid) {
    if (valid) {
        Af.q[0] = *(const v4u*)(ap);
        Af.q[1] = *(const v4u*)(ap + 8);
    } else {
        v4u z = {0u, 0u, 0u, 0u};
        Af.q[0] = z;
        Af.q[1] = z;
    }
}

// ---- Tensor Data Mover: 2D tile (rows x tileK f32, row stride strideElems) -> LDS ----
// D# packing per CDNA5 ISA 8.3/8.4: group0 {count=1, lds_addr, global_addr, type=2},
// group1 {data_size=4B, tensor_dim0=tileK, tensor_dim1=rows, tile_dim0=tileK,
//         tile_dim1=rows, tensor_dim0_stride=strideElems}. Groups 2/3 zero (2D tensor).
__device__ __forceinline__ void tdm_load_2d(unsigned lds_addr, const float* gptr,
                                            unsigned tileK, unsigned rows,
                                            unsigned long long strideElems) {
    unsigned long long ga = (unsigned long long)gptr;
    v4u g0;
    g0[0] = 1u;                                             // count=1, user D#
    g0[1] = lds_addr;                                       // LDS byte address
    g0[2] = (unsigned)(ga & 0xFFFFFFFFu);                   // global_addr[31:0]
    g0[3] = (unsigned)((ga >> 32) & 0x1FFFFFFu) | (2u << 30); // global_addr[56:32] | type=2
    v8i g1;
    g1[0] = (int)(2u << 16);                                // data_size=2 (4 bytes)
    g1[1] = (int)((tileK & 0xFFFFu) << 16);                 // tensor_dim0 lo16
    g1[2] = (int)((tileK >> 16) & 0xFFFFu) | (int)((rows & 0xFFFFu) << 16); // dim0 hi | dim1 lo
    g1[3] = (int)((rows >> 16) & 0xFFFFu) | (int)((tileK & 0xFFFFu) << 16); // dim1 hi | tile_dim0
    g1[4] = (int)(rows & 0xFFFFu);                          // tile_dim1 (tile_dim2=0)
    g1[5] = (int)(strideElems & 0xFFFFFFFFu);               // tensor_dim0_stride lo32
    g1[6] = (int)((strideElems >> 32) & 0xFFFFu);           // stride hi16 (dim1_stride=0)
    g1[7] = 0;
    v4i z = {0, 0, 0, 0};
#if __clang_major__ >= 23
    v8i z8 = {0, 0, 0, 0, 0, 0, 0, 0};
    __builtin_amdgcn_tensor_load_to_lds(g0, g1, z, z, z8, 0);
#else
    __builtin_amdgcn_tensor_load_to_lds(g0, g1, z, z, 0);
#endif
}

// ---------------- conv1 (4,1,256,256)->(4,32,256,256) + SELU ----------------
__global__ void conv1_selu_kernel(const float* __restrict__ x, const float* __restrict__ w,
                                  const float* __restrict__ bias, float* __restrict__ out) {
    long idx = (long)blockIdx.x * 256 + threadIdx.x;            // 4*32*65536
    int px = idx & 255, py = (idx >> 8) & 255, oc = (idx >> 16) & 31;
    int b = (int)(idx >> 21);
    const float* xin = x + (size_t)b * 65536;
    float s = bias[oc];
#pragma unroll
    for (int ky = 0; ky < 3; ++ky) {
        int iy = py + ky - 1;
        if (iy < 0 || iy > 255) continue;
#pragma unroll
        for (int kx = 0; kx < 3; ++kx) {
            int ix = px + kx - 1;
            if (ix < 0 || ix > 255) continue;
            s += xin[iy * 256 + ix] * w[oc * 9 + ky * 3 + kx];
        }
    }
    out[idx] = selu_f(s);
}

// ---------------- 2x2 maxpool (square) ----------------
__global__ void maxpool_kernel(const float* __restrict__ in, float* __restrict__ out,
                               int outHW, long total) {
    long idx = (long)blockIdx.x * 256 + threadIdx.x;
    if (idx >= total) return;
    int x = (int)(idx % outHW);
    int y = (int)((idx / outHW) % outHW);
    long n = idx / ((long)outHW * outHW);
    int inW = outHW * 2;
    const float* p = in + (size_t)n * inW * inW + (size_t)(2 * y) * inW + 2 * x;
    float m = fmaxf(fmaxf(p[0], p[1]), fmaxf(p[inW], p[inW + 1]));
    out[idx] = m;
}

// ---------------- conv2 (4,32,128,128)->(4,64,128,128) + SELU ----------------
__global__ void conv2_selu_kernel(const float* __restrict__ in, const float* __restrict__ w,
                                  const float* __restrict__ bias, float* __restrict__ out) {
    long idx = (long)blockIdx.x * 256 + threadIdx.x;            // 4*64*16384
    int px = idx & 127, py = (idx >> 7) & 127, oc = (idx >> 14) & 63;
    int b = (int)(idx >> 20);
    float s = bias[oc];
    for (int ic = 0; ic < 32; ++ic) {
        const float* xin = in + (size_t)(b * 32 + ic) * 16384;
        const float* wk = w + (size_t)(oc * 32 + ic) * 9;
#pragma unroll
        for (int ky = 0; ky < 3; ++ky) {
            int iy = py + ky - 1;
            if (iy < 0 || iy > 127) continue;
#pragma unroll
            for (int kx = 0; kx < 3; ++kx) {
                int ix = px + kx - 1;
                if (ix < 0 || ix > 127) continue;
                s += xin[iy * 128 + ix] * wk[ky * 3 + kx];
            }
        }
    }
    out[idx] = selu_f(s);
}

// ---------------- global average pool: (4,64,64,64) -> (4,64) ----------------
__global__ void gap_kernel(const float* __restrict__ h, float* __restrict__ gap) {
    __shared__ float sm[256];
    int bc = blockIdx.x;
    const float* p = h + (size_t)bc * 4096;
    float acc = 0.0f;
    for (int i = threadIdx.x; i < 4096; i += 256) acc += p[i];
    sm[threadIdx.x] = acc;
    __syncthreads();
    for (int st = 128; st > 0; st >>= 1) {
        if (threadIdx.x < st) sm[threadIdx.x] += sm[threadIdx.x + st];
        __syncthreads();
    }
    if (threadIdx.x == 0) gap[bc] = sm[0] * (1.0f / 4096.0f);
}

// ---------------- tiny SE MLP + temperature softmax ----------------
__global__ void att_mlp_kernel(const float* __restrict__ gap,
                               const float* __restrict__ w1, const float* __restrict__ b1,
                               const float* __restrict__ w2, const float* __restrict__ b2,
                               float* __restrict__ att) {
    __shared__ float sa[4][16];
    __shared__ float sl[4][64];
    int t = threadIdx.x;                                        // 64 threads
    if (t < 16) {
        for (int b = 0; b < 4; ++b) {
            float s = b1[t];
            for (int c = 0; c < 64; ++c) s += gap[b * 64 + c] * w1[t * 64 + c];
            sa[b][t] = fmaxf(s, 0.0f);
        }
    }
    __syncthreads();
    for (int b = 0; b < 4; ++b) {
        float s = b2[t];
#pragma unroll
        for (int j = 0; j < 16; ++j) s += sa[b][j] * w2[t * 16 + j];
        sl[b][t] = s * (1.0f / 34.0f);
    }
    __syncthreads();
    if (t < 4) {
        int b = t;
        float mx = -1e30f;
        for (int c = 0; c < 64; ++c) mx = fmaxf(mx, sl[b][c]);
        float sum = 0.0f;
        for (int c = 0; c < 64; ++c) sum += expf(sl[b][c] - mx);
        float inv = 1.0f / sum;
        for (int c = 0; c < 64; ++c) att[b * 64 + c] = expf(sl[b][c] - mx) * inv;
    }
}

// ---------------- apply channel attention in-place on (4,64,64,64) ----------------
__global__ void attn_scale_kernel(float* __restrict__ h, const float* __restrict__ att) {
    long idx = (long)blockIdx.x * 256 + threadIdx.x;            // 4*262144
    int b = (int)(idx >> 18);
    int ch = (int)((idx >> 12) & 63);
    h[idx] *= att[b * 64 + ch];
}

// ---------------- augmented activations: x -> silu(x) (base) and B0..B7(x) (spline), bf16 ----------------
__global__ void augment_kernel(const float* __restrict__ x,
                               unsigned short* __restrict__ Abase,
                               unsigned short* __restrict__ Aspl, long n) {
    long idx = (long)blockIdx.x * 256 + threadIdx.x;
    if (idx >= n) return;
    float v = x[idx];
    float sil = v / (1.0f + expf(-v));
    float bas[8];
    bspline8(v, bas);
    Abase[idx] = f2bf(sil);
    unsigned short* dst = Aspl + idx * 8;
#pragma unroll
    for (int c = 0; c < 8; ++c) dst[c] = f2bf(bas[c]);
}

// ---------------- KAN layer: TDM-staged spline contraction + direct base contraction ----------------
// spline: K = IN*8. TDM double-buffers 16-row x KCHUNK f32 weight tiles into LDS;
//         8 waves split each chunk's K=32 WMMA steps, B fragments read from LDS.
// base:   K = IN, direct streamed loads.
// grid.x = OUT/16 (N tiles), grid.y = KS (K slices), 256 threads = 8 waves.
// dynamic LDS = 2*16*KCHUNK*4 = 64KB (reduction scratch overlapped after pipeline drains).
__global__ void kan_wmma_kernel(const unsigned short* __restrict__ Abase,
                                const unsigned short* __restrict__ Aspl,
                                const float* __restrict__ base_w,
                                const float* __restrict__ spline_w,
                                const float* __restrict__ scaler,
                                float* __restrict__ partial,
                                long IN, int OUT, int KS) {
    extern __shared__ char smem[];
    float* tile = (float*)smem;                 // [2][16*KCHUNK]
    float* red = (float*)smem;                  // overlapped: used after spline+base loops

    const int wave = threadIdx.x >> 5, lane = threadIdx.x & 31;
    const int m = lane & 15, g = lane >> 4;     // A row (batch) / B col share lane&15
    const bool valid = (m < 4);
    const int o = blockIdx.x * 16 + m;

    const float* bw = base_w + (size_t)o * IN;
    const float* sc = scaler + (size_t)o * IN;
    const unsigned short* arB = Abase + (size_t)m * IN;
    const unsigned short* arS = Aspl + (size_t)m * IN * 8;

    v8f acc = {0.f, 0.f, 0.f, 0.f, 0.f, 0.f, 0.f, 0.f};

    // ---- spline contraction via Tensor Data Mover (8/9 of weight traffic) ----
    {
        const long sliceElems = (IN * 8) / KS;              // per-row K range of this block
        const int nchunk = (int)(sliceElems / KCHUNK);
        const long sliceBase = (long)blockIdx.y * sliceElems;
        const float* grow0 = spline_w + (size_t)(blockIdx.x * 16) * (IN * 8) + sliceBase;
        const unsigned long long strideElems = (unsigned long long)(IN * 8);

        if (wave == 0)
            tdm_load_2d((unsigned)(size_t)&tile[0], grow0, KCHUNK, 16, strideElems);

        for (int t = 0; t < nchunk; ++t) {
            if (wave == 0) {
                if (t + 1 < nchunk) {
                    tdm_load_2d((unsigned)(size_t)&tile[((t + 1) & 1) * 16 * KCHUNK],
                                grow0 + (long)(t + 1) * KCHUNK, KCHUNK, 16, strideElems);
                    __builtin_amdgcn_s_wait_tensorcnt(1);   // chunk t landed
                } else {
                    __builtin_amdgcn_s_wait_tensorcnt(0);
                }
            }
            __syncthreads();                                // tile[t&1] visible to all waves

            const float* lbuf = tile + (t & 1) * 16 * KCHUNK;
#pragma unroll
            for (int ss = 0; ss < KCHUNK / 32 / 8; ++ss) {  // 2 steps per wave
                const int s = wave * (KCHUNK / 32 / 8) + ss;
                const long k0 = sliceBase + (long)t * KCHUNK + s * 32;

                U16 Af;
                loadA(Af, arS + k0 + 16 * g, valid);

                const float* lp = lbuf + m * KCHUNK + s * 32 + 16 * g;
                v4f w0 = *(const v4f*)(lp);
                v4f w1 = *(const v4f*)(lp + 4);
                v4f w2 = *(const v4f*)(lp + 8);
                v4f w3 = *(const v4f*)(lp + 12);
                const long i0 = (k0 + 16 * g) >> 3;
                const float s0 = sc[i0], s1 = sc[i0 + 1];

                U16 Bf;
#pragma unroll
                for (int e = 0; e < 4; ++e) {
                    Bf.b[e]      = (__bf16)(w0[e] * s0);
                    Bf.b[4 + e]  = (__bf16)(w1[e] * s0);
                    Bf.b[8 + e]  = (__bf16)(w2[e] * s1);
                    Bf.b[12 + e] = (__bf16)(w3[e] * s1);
                }
                acc = __builtin_amdgcn_wmma_f32_16x16x32_bf16(false, Af.b, false, Bf.b,
                                                              (short)0, acc, false, false);
            }
            __syncthreads();                                // all reads done before buffer reuse
        }
    }

    // ---- base contraction, direct streamed loads (1/9 of weight traffic) ----
    {
        const long KT = IN >> 5;
        const long perBlock = KT / KS;
        const long perWave = perBlock >> 3;
        const long it0 = (long)blockIdx.y * perBlock + (long)wave * perWave;
        for (long it = 0; it < perWave; ++it) {
            const long k0 = (it0 + it) << 5;
            U16 Af;
            loadA(Af, arB + k0 + 16 * g, valid);

            const float* wp = bw + k0 + 16 * g;
            __builtin_prefetch((const void*)(wp + 4096), 0, 0);
            v4f w0 = *(const v4f*)(wp);
            v4f w1 = *(const v4f*)(wp + 4);
            v4f w2 = *(const v4f*)(wp + 8);
            v4f w3 = *(const v4f*)(wp + 12);

            U16 Bf;
#pragma unroll
            for (int e = 0; e < 4; ++e) {
                Bf.b[e]      = (__bf16)w0[e];
                Bf.b[4 + e]  = (__bf16)w1[e];
                Bf.b[8 + e]  = (__bf16)w2[e];
                Bf.b[12 + e] = (__bf16)w3[e];
            }
            acc = __builtin_amdgcn_wmma_f32_16x16x32_bf16(false, Af.b, false, Bf.b,
                                                          (short)0, acc, false, false);
        }
    }

    // ---- cross-wave reduction in LDS (deterministic); red overlaps tile (barriers order it) ----
    __syncthreads();
#pragma unroll
    for (int j = 0; j < 8; ++j) red[(wave * 32 + lane) * 8 + j] = acc[j];
    __syncthreads();
    if (wave == 0 && lane < 16) {
        // D layout: lanes 0-15 hold N=lane, VGPR r holds row M=r -> batch rows 0..3 are r=0..3
#pragma unroll
        for (int r = 0; r < 4; ++r) {
            float s = 0.0f;
#pragma unroll
            for (int w2i = 0; w2i < 8; ++w2i) s += red[(w2i * 32 + lane) * 8 + r];
            partial[((size_t)blockIdx.y * 4 + r) * OUT + blockIdx.x * 16 + lane] = s;
        }
    }
}

// ---------------- reduce K-slices (+ optional SELU) ----------------
__global__ void kan_reduce_kernel(const float* __restrict__ partial, float* __restrict__ out,
                                  int OUT, int KS, int applySelu) {
    long idx = (long)blockIdx.x * 256 + threadIdx.x;            // over 4*OUT
    if (idx >= 4L * OUT) return;
    int b = (int)(idx / OUT), o = (int)(idx % OUT);
    float s = 0.0f;
    for (int k = 0; k < KS; ++k) s += partial[((size_t)k * 4 + b) * OUT + o];
    out[idx] = applySelu ? selu_f(s) : s;
}

// ---------------- bilinear upsample rows: (4,64,16,16) -> (4,64,256,16) ----------------
__global__ void up_rows_kernel(const float* __restrict__ v, float* __restrict__ out) {
    long idx = (long)blockIdx.x * 256 + threadIdx.x;            // 4*64*256*16
    int w = idx & 15, H = (idx >> 4) & 255;
    long bc = idx >> 12;                                        // b*64+c
    float pos = H * (15.0f / 255.0f);
    int lo = (int)floorf(pos);
    if (lo > 14) lo = 14;
    float f = pos - (float)lo;
    const float* p = v + (size_t)bc * 256;
    out[idx] = (1.0f - f) * p[lo * 16 + w] + f * p[(lo + 1) * 16 + w];
}

// ---------------- bilinear upsample cols: (4,64,256,16) -> (4,64,256,256) ----------------
__global__ void up_cols_kernel(const float* __restrict__ v, float* __restrict__ out) {
    long idx = (long)blockIdx.x * 256 + threadIdx.x;            // 4*64*256*256
    int X = idx & 255;
    long rowid = idx >> 8;                                      // (b*64+c)*256+H
    float pos = X * (15.0f / 255.0f);
    int lo = (int)floorf(pos);
    if (lo > 14) lo = 14;
    float f = pos - (float)lo;
    const float* p = v + (size_t)rowid * 16;
    out[idx] = (1.0f - f) * p[lo] + f * p[lo + 1];
}

// ---------------- final conv: (4,64,256,256) -> (4,1,256,256) ----------------
__global__ void final_conv_kernel(const float* __restrict__ in, const float* __restrict__ w,
                                  const float* __restrict__ bias, float* __restrict__ out) {
    long idx = (long)blockIdx.x * 256 + threadIdx.x;            // 4*65536
    int px = idx & 255, py = (idx >> 8) & 255;
    int b = (int)(idx >> 16);
    float s = bias[0];
    for (int c = 0; c < 64; ++c) {
        const float* xin = in + (size_t)(b * 64 + c) * 65536;
        const float* wk = w + (size_t)c * 9;
#pragma unroll
        for (int ky = 0; ky < 3; ++ky) {
            int iy = py + ky - 1;
            if (iy < 0 || iy > 255) continue;
#pragma unroll
            for (int kx = 0; kx < 3; ++kx) {
                int ix = px + kx - 1;
                if (ix < 0 || ix > 255) continue;
                s += xin[iy * 256 + ix] * wk[ky * 3 + kx];
            }
        }
    }
    out[idx] = s;
}

extern "C" void kernel_launch(void* const* d_in, const int* in_sizes, int n_in,
                              void* d_out, int out_size, void* d_ws, size_t ws_size,
                              hipStream_t stream) {
    const float* x        = (const float*)d_in[0];
    const float* conv1_w  = (const float*)d_in[1];
    const float* conv1_b  = (const float*)d_in[2];
    const float* conv2_w  = (const float*)d_in[3];
    const float* conv2_b  = (const float*)d_in[4];
    const float* att1_w   = (const float*)d_in[5];
    const float* att1_b   = (const float*)d_in[6];
    const float* att2_w   = (const float*)d_in[7];
    const float* att2_b   = (const float*)d_in[8];
    const float* k1_base  = (const float*)d_in[9];
    const float* k1_spl   = (const float*)d_in[10];
    const float* k1_scal  = (const float*)d_in[11];
    const float* k2_base  = (const float*)d_in[12];
    const float* k2_spl   = (const float*)d_in[13];
    const float* k2_scal  = (const float*)d_in[14];
    const float* final_w  = (const float*)d_in[15];
    const float* final_b  = (const float*)d_in[16];
    float* out = (float*)d_out;

    char* ws = (char*)d_ws;
    float*          h1   = (float*)(ws + 0);                    // 33554432 B
    float*          p1   = (float*)(ws + 33554432);             // 16777216 B
    float*          h2   = (float*)(ws + 50331648);             // 16777216 B
    float*          p2   = (float*)(ws + 67108864);             //  4194304 B
    float*          gap  = (float*)(ws + 71303168);             //     1024 B
    float*          att  = (float*)(ws + 71304192);             //     1024 B
    unsigned short* A1b  = (unsigned short*)(ws + 71305216);    //  2097152 B
    unsigned short* A1s  = (unsigned short*)(ws + 73402368);    // 16777216 B
    float*          o1   = (float*)(ws + 90179584);             //    16384 B
    unsigned short* A2b  = (unsigned short*)(ws + 90195968);    //     8192 B
    unsigned short* A2s  = (unsigned short*)(ws + 90204160);    //    65536 B
    float*          o2   = (float*)(ws + 90269696);             //   262144 B
    float*          up1  = (float*)(ws + 90531840);             //  4194304 B
    float*          up2  = (float*)(ws + 94726144);             // 67108864 B
    float*          P1   = (float*)(ws + 161835008);            //   524288 B
    float*          P2   = (float*)(ws + 162359296);            //  1048576 B

    const long IN1 = 262144, IN2 = 1024;
    const int  OUT1 = 1024, OUT2 = 16384;
    const int  KS1 = 32, KS2 = 4;
    const unsigned ldsBytes = 2u * 16u * KCHUNK * 4u;           // 64KB double buffer

    conv1_selu_kernel<<<32768, 256, 0, stream>>>(x, conv1_w, conv1_b, h1);
    maxpool_kernel<<<8192, 256, 0, stream>>>(h1, p1, 128, 4L * 32 * 128 * 128);
    conv2_selu_kernel<<<16384, 256, 0, stream>>>(p1, conv2_w, conv2_b, h2);
    maxpool_kernel<<<4096, 256, 0, stream>>>(h2, p2, 64, 4L * 64 * 64 * 64);

    gap_kernel<<<256, 256, 0, stream>>>(p2, gap);
    att_mlp_kernel<<<1, 64, 0, stream>>>(gap, att1_w, att1_b, att2_w, att2_b, att);
    attn_scale_kernel<<<4096, 256, 0, stream>>>(p2, att);

    augment_kernel<<<4096, 256, 0, stream>>>(p2, A1b, A1s, 4L * IN1);
    {
        dim3 g((unsigned)(OUT1 / 16), (unsigned)KS1);
        kan_wmma_kernel<<<g, 256, ldsBytes, stream>>>(A1b, A1s, k1_base, k1_spl, k1_scal, P1, IN1, OUT1, KS1);
    }
    kan_reduce_kernel<<<16, 256, 0, stream>>>(P1, o1, OUT1, KS1, 1);

    augment_kernel<<<16, 256, 0, stream>>>(o1, A2b, A2s, 4L * IN2);
    {
        dim3 g((unsigned)(OUT2 / 16), (unsigned)KS2);
        kan_wmma_kernel<<<g, 256, ldsBytes, stream>>>(A2b, A2s, k2_base, k2_spl, k2_scal, P2, IN2, OUT2, KS2);
    }
    kan_reduce_kernel<<<256, 256, 0, stream>>>(P2, o2, OUT2, KS2, 0);

    up_rows_kernel<<<4096, 256, 0, stream>>>(o2, up1);
    up_cols_kernel<<<65536, 256, 0, stream>>>(up1, up2);
    final_conv_kernel<<<1024, 256, 0, stream>>>(up2, final_w, final_b, out);
}
